// WMultiHeadAttention_38757784879163
// MI455X (gfx1250) — matrix operand order repaired and checked by reading, not compiled, not run on hardware
//
#include <hip/hip_runtime.h>

#define B_    2
#define N_    4096
#define E_    512
#define H_    8
#define D_    64
#define WHALF 128

#ifndef __has_builtin
#define __has_builtin(x) 0
#endif
#if __has_builtin(__builtin_amdgcn_tensor_load_to_lds) && \
    __has_builtin(__builtin_amdgcn_s_wait_tensorcnt)
#define HAVE_TDM 1
#else
#define HAVE_TDM 0
#endif

typedef _Float16 h16;
typedef __attribute__((ext_vector_type(16))) _Float16 v16h;
typedef __attribute__((ext_vector_type(8)))  _Float16 v8h;
typedef __attribute__((ext_vector_type(8)))  float    v8f;
typedef __attribute__((ext_vector_type(4)))  unsigned int v4u;
typedef __attribute__((ext_vector_type(8)))  int      v8i;
typedef __attribute__((ext_vector_type(4)))  int      v4i;

union V16U { v16h v; v8h h[2]; };

__device__ __forceinline__ v8f wmma16(v16h a, v16h b, v8f c) {
  return __builtin_amdgcn_wmma_f32_16x16x32_f16(false, a, false, b, (short)0, c,
                                                false, false);
}

// A fragment: rows r0..r0+15, contraction k0..k0+31 from row-major f16 (ld halves).
// Lane m=lane&15 holds row r0+m; e=0..7 -> K=k0+8*hi+e, e=8..15 -> K=k0+16+8*hi+(e-8).
__device__ __forceinline__ v16h frag_a(const h16* p, int ld, int r0, int k0) {
  const int lane = threadIdx.x & 31;
  const int hi = lane >> 4, m = lane & 15;
  const h16* row = p + (size_t)(r0 + m) * ld + (k0 + hi * 8);
  V16U u;
  u.h[0] = *(const v8h*)row;
  u.h[1] = *(const v8h*)(row + 16);
  return u.v;
}

// B fragment: contraction k0..k0+31, cols c0..c0+15, sourced from B^T row-major:
// B[k][c] = p[c*ld + k].  Lane n=lane&15 holds col c0+n; e=0..7 -> K=k0+8*hi+e, e=8..15 -> K=k0+16+8*hi+(e-8).
__device__ __forceinline__ v16h frag_b(const h16* p, int ld, int c0, int k0) {
  const int lane = threadIdx.x & 31;
  const int hi = lane >> 4, n = lane & 15;
  const h16* row = p + (size_t)(c0 + n) * ld + (k0 + hi * 8);
  V16U u;
  u.h[0] = *(const v8h*)row;
  u.h[1] = *(const v8h*)(row + 16);
  return u.v;
}

// ---------------- TDM: 1-D async copy global->LDS with OOB zero-fill ----------
__device__ __forceinline__ void tdm_load_1d(const h16* gsrc, h16* ldst,
                                            int tile_elems, int tensor_elems) {
#if HAVE_TDM
  unsigned long long ga = (unsigned long long)(uintptr_t)gsrc;
  unsigned lds = (unsigned)(uintptr_t)ldst;   // low 32 bits = LDS byte offset
  v4u g0;
  g0[0] = 1u;                                             // count=1, user D#
  g0[1] = lds;                                            // lds_addr
  g0[2] = (unsigned)ga;                                   // global_addr[31:0]
  g0[3] = (unsigned)((ga >> 32) & 0x1FFFFFFull) | (2u << 30);  // addr[56:32], type=2
  v8i g1;
  g1[0] = 0x00010000;                                     // data_size = 2 bytes
  g1[1] = (int)(((unsigned)tensor_elems & 0xFFFFu) << 16);        // tensor_dim0 lo
  g1[2] = (int)((((unsigned)tensor_elems >> 16) & 0xFFFFu) | (1u << 16)); // hi | dim1=1
  g1[3] = (int)(((unsigned)tile_elems & 0xFFFFu) << 16);  // tile_dim0
  g1[4] = 0;                                              // tile_dim1/2 = 0 (1-D)
  g1[5] = tile_elems;                                     // dim0 stride (unused, 1-D)
  g1[6] = 0;
  g1[7] = 0;
  v4i z4 = {0, 0, 0, 0};
#if __clang_major__ >= 23
  v8i z8 = {0, 0, 0, 0, 0, 0, 0, 0};
  __builtin_amdgcn_tensor_load_to_lds(g0, g1, z4, z4, z8, 0);
#else
  __builtin_amdgcn_tensor_load_to_lds(g0, g1, z4, z4, 0);
#endif
#else
  (void)gsrc; (void)ldst; (void)tile_elems; (void)tensor_elems;
#endif
}

// ---------------- prep: f32 -> f16 conversions ----------------
__global__ __launch_bounds__(256) void prep_x(const float* __restrict__ x,
                                              h16* __restrict__ xh, int n8) {
  int i = blockIdx.x * 256 + threadIdx.x;
  if (i >= n8) return;
  const float4* p = (const float4*)x + (size_t)i * 2;
  float4 a = p[0], b = p[1];
  union { v8h v; h16 e[8]; } u;
  u.e[0] = (h16)a.x; u.e[1] = (h16)a.y; u.e[2] = (h16)a.z; u.e[3] = (h16)a.w;
  u.e[4] = (h16)b.x; u.e[5] = (h16)b.y; u.e[6] = (h16)b.z; u.e[7] = (h16)b.w;
  ((v8h*)xh)[i] = u.v;
}

__global__ __launch_bounds__(256) void prep_wt(const float* __restrict__ Wq,
                                               const float* __restrict__ Wk,
                                               const float* __restrict__ Wv,
                                               const float* __restrict__ Wfc,
                                               h16* __restrict__ WqT,
                                               h16* __restrict__ WkT,
                                               h16* __restrict__ WvT,
                                               h16* __restrict__ WfT) {
  const float* W; h16* T;
  switch (blockIdx.y) {
    case 0:  W = Wq;  T = WqT; break;
    case 1:  W = Wk;  T = WkT; break;
    case 2:  W = Wv;  T = WvT; break;
    default: W = Wfc; T = WfT; break;
  }
  int i = blockIdx.x * 256 + threadIdx.x;   // over E_*E_
  int f = i >> 9, k = i & 511;
  T[(size_t)f * E_ + k] = (h16)W[(size_t)k * E_ + f];
}

// Ping-pong pipelined 16x64-strip GEMM body: acc += A[row0,:512] * B^T.
// Two fragment register sets alternate; no rotation copies.
__device__ __forceinline__ void gemm_strip(const h16* __restrict__ A,
                                           const h16* __restrict__ BT,
                                           int row0, int f0, v8f acc[4]) {
  v16h aA = frag_a(A, E_, row0, 0);
  v16h bA[4];
#pragma unroll
  for (int dg = 0; dg < 4; ++dg) bA[dg] = frag_b(BT, E_, f0 + dg * 16, 0);
  v16h aB;
  v16h bB[4];
  for (int ks = 32; ks < E_ - 32; ks += 64) {
    aB = frag_a(A, E_, row0, ks);
#pragma unroll
    for (int dg = 0; dg < 4; ++dg) bB[dg] = frag_b(BT, E_, f0 + dg * 16, ks);
#pragma unroll
    for (int dg = 0; dg < 4; ++dg) acc[dg] = wmma16(aA, bA[dg], acc[dg]);
    aA = frag_a(A, E_, row0, ks + 32);
#pragma unroll
    for (int dg = 0; dg < 4; ++dg) bA[dg] = frag_b(BT, E_, f0 + dg * 16, ks + 32);
#pragma unroll
    for (int dg = 0; dg < 4; ++dg) acc[dg] = wmma16(aB, bB[dg], acc[dg]);
  }
  aB = frag_a(A, E_, row0, E_ - 32);
#pragma unroll
  for (int dg = 0; dg < 4; ++dg) bB[dg] = frag_b(BT, E_, f0 + dg * 16, E_ - 32);
#pragma unroll
  for (int dg = 0; dg < 4; ++dg) acc[dg] = wmma16(aA, bA[dg], acc[dg]);
#pragma unroll
  for (int dg = 0; dg < 4; ++dg) acc[dg] = wmma16(aB, bB[dg], acc[dg]);
}

// ---------------- QKV projection GEMM ----------------
// Q,K stored [B,H,N,64]; V stored transposed [B,H,64,N].
__global__ __launch_bounds__(128) void qkv_gemm(
    const h16* __restrict__ xh,
    const h16* __restrict__ WqT, const h16* __restrict__ WkT,
    const h16* __restrict__ WvT,
    const float* __restrict__ bq, const float* __restrict__ bk,
    const float* __restrict__ bv,
    h16* __restrict__ Qh, h16* __restrict__ Kh, h16* __restrict__ Vt) {
  const int which = blockIdx.z;
  const h16* WT = (which == 0) ? WqT : (which == 1) ? WkT : WvT;
  const float* bias = (which == 0) ? bq : (which == 1) ? bk : bv;
  const int wave = threadIdx.x >> 5, lane = threadIdx.x & 31;
  const int hi = lane >> 4, nl = lane & 15;
  const int row0 = blockIdx.x * 64 + wave * 16;   // flat row in [0, B*N)
  const int f0 = blockIdx.y * 64;

  v8f acc[4] = {};
  gemm_strip(xh, WT, row0, f0, acc);

  const int bz = row0 / N_;
  const int n0 = row0 % N_;
  if (which == 2) {
#pragma unroll
    for (int dg = 0; dg < 4; ++dg) {
      int f = f0 + dg * 16 + nl;
      float bval = bias[f];
      int hh = f >> 6, d = f & 63;
      h16* dst = Vt + (((size_t)bz * H_ + hh) * D_ + d) * N_;
#pragma unroll
      for (int r = 0; r < 8; ++r)
        dst[n0 + r + 8 * hi] = (h16)(acc[dg][r] + bval);
    }
  } else {
    h16* O = (which == 1) ? Kh : Qh;
#pragma unroll
    for (int dg = 0; dg < 4; ++dg) {
      int f = f0 + dg * 16 + nl;
      float bval = bias[f];
      int hh = f >> 6, d = f & 63;
#pragma unroll
      for (int r = 0; r < 8; ++r)
        O[(((size_t)bz * H_ + hh) * N_ + n0 + r + 8 * hi) * D_ + d] =
            (h16)(acc[dg][r] + bval);
    }
  }
}

// ---------------- windowed flash attention ----------------
// 4 waves/block; wave owns 16 query rows; streams 32-key tiles over |i-j|<=128.
// K tiles are TDM-prefetched (double-buffered) into per-wave LDS; TDM zero-fills
// the sequence tail.  V fragments come straight from global (per-lane contiguous).
__global__ __launch_bounds__(128) void attn_win(const h16* __restrict__ Qh,
                                                const h16* __restrict__ Kh,
                                                const h16* __restrict__ Vt,
                                                h16* __restrict__ Oh) {
  __shared__ h16 Kbuf[4][2][32 * D_];   // 32 KB
  __shared__ h16 Pbuf[4][16 * 32];      // 4 KB
  const int wave = threadIdx.x >> 5, lane = threadIdx.x & 31;
  const int hi = lane >> 4, nl = lane & 15;
  const int bz = blockIdx.z, hh = blockIdx.y;
  const int q0 = blockIdx.x * 64 + wave * 16;

  const h16* Qp = Qh + ((size_t)bz * H_ + hh) * N_ * D_;
  const h16* Kp = Kh + ((size_t)bz * H_ + hh) * N_ * D_;
  const h16* Vp = Vt + ((size_t)bz * H_ + hh) * D_ * N_;
  h16* Pw = Pbuf[wave];

  const v16h aQ0 = frag_a(Qp, D_, q0, 0);
  const v16h aQ1 = frag_a(Qp, D_, q0, 32);

  v8f o[4] = {};
  float mrun[8], lrun[8];
#pragma unroll
  for (int r = 0; r < 8; ++r) { mrun[r] = -1e30f; lrun[r] = 0.0f; }
  const float scale = 0.125f;  // 1/sqrt(64)

  int jlo = q0 - WHALF; if (jlo < 0) jlo = 0;
  int jhi = q0 + 15 + WHALF; if (jhi > N_ - 1) jhi = N_ - 1;
  const int jt0 = jlo & ~31;
  const int ntiles = (jhi - jt0) / 32 + 1;

#if HAVE_TDM
  tdm_load_1d(Kp + (size_t)jt0 * D_, Kbuf[wave][0], 32 * D_, (N_ - jt0) * D_);
#endif
  int cur = 0;

  for (int t = 0; t < ntiles; ++t) {
    const int jt = jt0 + t * 32;
#if HAVE_TDM
    if (t + 1 < ntiles) {
      int jn = jt + 32;
      tdm_load_1d(Kp + (size_t)jn * D_, Kbuf[wave][cur ^ 1], 32 * D_,
                  (N_ - jn) * D_);
      __builtin_amdgcn_s_wait_tensorcnt(1);   // older in-order transfer done
    } else {
      __builtin_amdgcn_s_wait_tensorcnt(0);
    }
    const h16* Kt = Kbuf[wave][cur];
    const int jb = 0;                          // key base within tile
#else
    const h16* Kt = Kp;
    const int jb = jt;
#endif
    // S = Q * K^T for 16q x 32k
    v8f s[2];
#pragma unroll
    for (int g = 0; g < 2; ++g) {
      v16h bk0 = frag_b(Kt, D_, jb + 16 * g, 0);
      v16h bk1 = frag_b(Kt, D_, jb + 16 * g, 32);
      v8f c = {};
      c = wmma16(aQ0, bk0, c);
      c = wmma16(aQ1, bk1, c);
      s[g] = c;
    }
    // hoist V fragment loads so they overlap the softmax VALU work
    v16h bv[4];
#pragma unroll
    for (int dg = 0; dg < 4; ++dg) bv[dg] = frag_b(Vp, N_, 16 * dg, jt);

    // scale + window mask, per-row tile max
    float tmax[8];
#pragma unroll
    for (int r = 0; r < 8; ++r) tmax[r] = -1e30f;
#pragma unroll
    for (int g = 0; g < 2; ++g)
#pragma unroll
      for (int r = 0; r < 8; ++r) {
        int q = q0 + r + 8 * hi;
        int j = jt + 16 * g + nl;
        bool in = (j >= q - WHALF) && (j <= q + WHALF) && (j < N_);
        float sv = in ? s[g][r] * scale : -1e30f;
        s[g][r] = sv;
        tmax[r] = fmaxf(tmax[r], sv);
      }
#pragma unroll
    for (int r = 0; r < 8; ++r) {
      float v = tmax[r];
      v = fmaxf(v, __shfl_xor(v, 1, 32));
      v = fmaxf(v, __shfl_xor(v, 2, 32));
      v = fmaxf(v, __shfl_xor(v, 4, 32));
      v = fmaxf(v, __shfl_xor(v, 8, 32));
      tmax[r] = v;
    }
    float alpha[8];
#pragma unroll
    for (int r = 0; r < 8; ++r) {
      float mn = fmaxf(mrun[r], tmax[r]);
      alpha[r] = __expf(mrun[r] - mn);
      mrun[r] = mn;
    }
    // P = exp(S - m); masked lanes exactly 0
    float tsum[8];
#pragma unroll
    for (int r = 0; r < 8; ++r) tsum[r] = 0.0f;
#pragma unroll
    for (int g = 0; g < 2; ++g)
#pragma unroll
      for (int r = 0; r < 8; ++r) {
        int q = q0 + r + 8 * hi;
        int j = jt + 16 * g + nl;
        bool in = (j >= q - WHALF) && (j <= q + WHALF) && (j < N_);
        float p = in ? __expf(s[g][r] - mrun[r]) : 0.0f;
        s[g][r] = p;
        tsum[r] += p;
      }
#pragma unroll
    for (int r = 0; r < 8; ++r) {
      float v = tsum[r];
      v += __shfl_xor(v, 1, 32);
      v += __shfl_xor(v, 2, 32);
      v += __shfl_xor(v, 4, 32);
      v += __shfl_xor(v, 8, 32);
      lrun[r] = lrun[r] * alpha[r] + v;
    }
    // rescale O accumulators
#pragma unroll
    for (int dg = 0; dg < 4; ++dg)
#pragma unroll
      for (int r = 0; r < 8; ++r) o[dg][r] *= alpha[r];
    // P: C-layout -> A-layout via per-wave LDS tile (row-major 16x32)
#pragma unroll
    for (int g = 0; g < 2; ++g)
#pragma unroll
      for (int r = 0; r < 8; ++r)
        Pw[(r + 8 * hi) * 32 + 16 * g + nl] = (h16)s[g][r];
    v16h aP = frag_a(Pw, 32, 0, 0);
    // O += P * V
#pragma unroll
    for (int dg = 0; dg < 4; ++dg) o[dg] = wmma16(aP, bv[dg], o[dg]);
    cur ^= 1;
  }
  // normalize + store O as f16 [B,N,H*64]
  float inv[8];
#pragma unroll
  for (int r = 0; r < 8; ++r) inv[r] = 1.0f / lrun[r];
#pragma unroll
  for (int dg = 0; dg < 4; ++dg)
#pragma unroll
    for (int r = 0; r < 8; ++r)
      Oh[((size_t)bz * N_ + q0 + r + 8 * hi) * E_ + hh * D_ + 16 * dg + nl] =
          (h16)(o[dg][r] * inv[r]);
}

// ---------------- output projection ----------------
__global__ __launch_bounds__(128) void out_gemm(const h16* __restrict__ Oh,
                                                const h16* __restrict__ WfT,
                                                const float* __restrict__ bfc,
                                                float* __restrict__ out) {
  const int wave = threadIdx.x >> 5, lane = threadIdx.x & 31;
  const int hi = lane >> 4, nl = lane & 15;
  const int row0 = blockIdx.x * 64 + wave * 16;   // flat row in [0, B*N)
  const int f0 = blockIdx.y * 64;

  v8f acc[4] = {};
  gemm_strip(Oh, WfT, row0, f0, acc);

#pragma unroll
  for (int dg = 0; dg < 4; ++dg) {
    int f = f0 + dg * 16 + nl;
    float bval = bfc[f];
#pragma unroll
    for (int r = 0; r < 8; ++r)
      out[(size_t)(row0 + r + 8 * hi) * E_ + f] = acc[dg][r] + bval;
  }
}

extern "C" void kernel_launch(void* const* d_in, const int* in_sizes, int n_in,
                              void* d_out, int out_size, void* d_ws, size_t ws_size,
                              hipStream_t stream) {
  const float* x   = (const float*)d_in[0];
  const float* Wq  = (const float*)d_in[1];
  const float* bq  = (const float*)d_in[2];
  const float* Wk  = (const float*)d_in[3];
  const float* bk  = (const float*)d_in[4];
  const float* Wv  = (const float*)d_in[5];
  const float* bv  = (const float*)d_in[6];
  const float* Wfc = (const float*)d_in[7];
  const float* bfc = (const float*)d_in[8];
  float* out = (float*)d_out;

  char* ws = (char*)d_ws;
  size_t off = 0;
  h16* xh  = (h16*)(ws + off); off += (size_t)B_ * N_ * E_ * 2;
  h16* WqT = (h16*)(ws + off); off += (size_t)E_ * E_ * 2;
  h16* WkT = (h16*)(ws + off); off += (size_t)E_ * E_ * 2;
  h16* WvT = (h16*)(ws + off); off += (size_t)E_ * E_ * 2;
  h16* WfT = (h16*)(ws + off); off += (size_t)E_ * E_ * 2;
  h16* Qh  = (h16*)(ws + off); off += (size_t)B_ * N_ * E_ * 2;
  h16* Kh  = (h16*)(ws + off); off += (size_t)B_ * N_ * E_ * 2;
  h16* Vt  = (h16*)(ws + off); off += (size_t)B_ * N_ * E_ * 2;
  h16* Oh  = (h16*)(ws + off); off += (size_t)B_ * N_ * E_ * 2;
  (void)ws_size; (void)in_sizes; (void)n_in; (void)out_size;

  prep_x<<<(B_ * N_ * E_ / 8 + 255) / 256, 256, 0, stream>>>(x, xh, B_ * N_ * E_ / 8);
  prep_wt<<<dim3(E_ * E_ / 256, 4, 1), 256, 0, stream>>>(Wq, Wk, Wv, Wfc,
                                                         WqT, WkT, WvT, WfT);
  qkv_gemm<<<dim3(B_ * N_ / 64, E_ / 64, 3), 128, 0, stream>>>(
      xh, WqT, WkT, WvT, bq, bk, bv, Qh, Kh, Vt);
  attn_win<<<dim3(N_ / 64, H_, B_), 128, 0, stream>>>(Qh, Kh, Vt, Oh);
  out_gemm<<<dim3(B_ * N_ / 64, E_ / 64, 1), 128, 0, stream>>>(Oh, WfT, bfc, out);
}
